// TradeFlowGAT_10668698764073
// MI455X (gfx1250) — matrix operand order repaired and checked by reading, not compiled, not run on hardware
//
#include <hip/hip_runtime.h>

// ---------------------------------------------------------------------------
// Types for CDNA5 WMMA (wave32)
// ---------------------------------------------------------------------------
typedef __attribute__((ext_vector_type(16))) __bf16 v16bf;
typedef __attribute__((ext_vector_type(8)))  __bf16 v8bf;
typedef __attribute__((ext_vector_type(4)))  __bf16 v4bf;
typedef __attribute__((ext_vector_type(8)))  float  v8f;
typedef __attribute__((ext_vector_type(4)))  float  v4f;

#define FK 544           // 518 features padded to 17 * 32
#define NEG_SLOPE 0.2f

// Load a 16-element bf16 fragment from two 8-element (16B) runs.
__device__ inline v16bf ldfrag2(const __bf16* lo, const __bf16* hi) {
    v8bf l = *(const v8bf*)lo;
    v8bf h = *(const v8bf*)hi;
    v16bf r;
#pragma unroll
    for (int i = 0; i < 8; ++i) { r[i] = l[i]; r[i + 8] = h[i]; }
    return r;
}

__device__ inline v8f wmma_bf16(v16bf a, v16bf b, v8f c) {
    return __builtin_amdgcn_wmma_f32_16x16x32_bf16(
        /*neg_a=*/false, a, /*neg_b=*/false, b,
        /*c_mod=*/(short)0, c, /*reuse_a=*/false, /*reuse_b=*/false);
}

// Monotone uint encoding of float for atomicMax-based float max.
__device__ inline unsigned fkey(float f) {
    unsigned u = __float_as_uint(f);
    return (u & 0x80000000u) ? ~u : (u | 0x80000000u);
}
__device__ inline float fdecode(unsigned k) {
    unsigned u = (k & 0x80000000u) ? (k & 0x7fffffffu) : ~k;
    return __uint_as_float(u);
}
#define NEG_INF_KEY 0x007FFFFFu   // fkey(-inf)

// ---------------------------------------------------------------------------
// Prep: transpose fp32 [R x C] weight to bf16 [C x padK] (row = out column)
// ---------------------------------------------------------------------------
__global__ void tr_bf16(const float* __restrict__ src, int R, int C,
                        __bf16* __restrict__ dst, int padK) {
    int idx = blockIdx.x * blockDim.x + threadIdx.x;
    int total = C * padK;
    if (idx >= total) return;
    int n = idx / padK, k = idx % padK;
    dst[idx] = (k < R) ? (__bf16)src[(size_t)k * C + n] : (__bf16)0.0f;
}

// ---------------------------------------------------------------------------
// Layer-0 node transforms (K = 3): xl0, xr0, res0
// ---------------------------------------------------------------------------
__global__ void lin0(const float* __restrict__ x,
                     const float* __restrict__ Wl, const float* __restrict__ bl,
                     const float* __restrict__ Wr, const float* __restrict__ br,
                     const float* __restrict__ Rw, const float* __restrict__ Rb,
                     float* __restrict__ xl, float* __restrict__ xr,
                     float* __restrict__ res, int N_) {
    long long idx = (long long)blockIdx.x * blockDim.x + threadIdx.x;
    if (idx >= (long long)N_ * 128) return;
    int n = (int)(idx >> 7), c = (int)(idx & 127);
    float x0 = x[n * 3 + 0], x1 = x[n * 3 + 1], x2 = x[n * 3 + 2];
    xl[idx]  = x0 * Wl[c] + x1 * Wl[128 + c] + x2 * Wl[256 + c] + bl[c];
    xr[idx]  = x0 * Wr[c] + x1 * Wr[128 + c] + x2 * Wr[256 + c] + br[c];
    res[idx] = x0 * Rw[c] + x1 * Rw[128 + c] + x2 * Rw[256 + c] + Rb[c];
}

// ---------------------------------------------------------------------------
// Zero segment accumulators, set max keys to -inf
// ---------------------------------------------------------------------------
__global__ void init_seg(float* __restrict__ den, unsigned* __restrict__ mx,
                         float* __restrict__ agg, int N_) {
    long long idx = (long long)blockIdx.x * blockDim.x + threadIdx.x;
    if (idx < (long long)N_ * 128) agg[idx] = 0.0f;
    if (idx < (long long)N_ * 4) { den[idx] = 0.0f; mx[idx] = NEG_INF_KEY; }
}

// ---------------------------------------------------------------------------
// Edge attention scores + segment max (wave per edge, lane = channel)
// Weights staged global->LDS with CDNA5 async-to-LDS (ASYNCcnt tracked).
// ---------------------------------------------------------------------------
__global__ void edge_score(const int* __restrict__ src, const int* __restrict__ dst,
                           const float* __restrict__ ea,
                           const float* __restrict__ xl, const float* __restrict__ xr,
                           const float* __restrict__ We, const float* __restrict__ att,
                           float* __restrict__ s_out, unsigned* __restrict__ mx,
                           long long E_) {
    __shared__ __align__(16) float sWe[6 * 128];   // 192 float4 chunks
    __shared__ __align__(16) float sAtt[128];      //  32 float4 chunks
    int tid = threadIdx.x;
    if (tid < 224) {
        const float* gsrc = (tid < 192) ? (We + tid * 4) : (att + (tid - 192) * 4);
        float* ldst = (tid < 192) ? (sWe + tid * 4) : (sAtt + (tid - 192) * 4);
        unsigned loff = (unsigned)(size_t)(void*)ldst;   // low 32 bits = LDS offset
        asm volatile("global_load_async_to_lds_b128 %0, %1, off"
                     :: "v"(loff), "v"(gsrc) : "memory");
    }
    asm volatile("s_wait_asynccnt 0x0" ::: "memory");
    __syncthreads();

    long long e = (long long)blockIdx.x * (blockDim.x >> 5) + (tid >> 5);
    if (e >= E_) return;
    int lane = tid & 31;
    int sN = src[e], dN = dst[e];
    float a0 = ea[e * 6 + 0], a1 = ea[e * 6 + 1], a2 = ea[e * 6 + 2];
    float a3 = ea[e * 6 + 3], a4 = ea[e * 6 + 4], a5 = ea[e * 6 + 5];

    float sv[4];
#pragma unroll
    for (int h = 0; h < 4; ++h) {
        int c = h * 32 + lane;
        float m = xl[(long long)sN * 128 + c] + xr[(long long)dN * 128 + c]
                + a0 * sWe[c] + a1 * sWe[128 + c] + a2 * sWe[256 + c]
                + a3 * sWe[384 + c] + a4 * sWe[512 + c] + a5 * sWe[640 + c];
        m = (m > 0.0f) ? m : NEG_SLOPE * m;   // leaky_relu
        float v = m * sAtt[c];
#pragma unroll
        for (int off = 16; off > 0; off >>= 1) v += __shfl_xor(v, off, 32);
        sv[h] = v;
    }
    if (lane == 0) {
#pragma unroll
        for (int h = 0; h < 4; ++h) {
            s_out[e * 4 + h] = sv[h];
            atomicMax(&mx[(long long)dN * 4 + h], fkey(sv[h]));
        }
    }
}

// ---------------------------------------------------------------------------
// exp(s - max), accumulate denominator and unnormalized aggregation
// ---------------------------------------------------------------------------
__global__ void edge_soft_agg(const int* __restrict__ src, const int* __restrict__ dst,
                              const float* __restrict__ s_in, const unsigned* __restrict__ mx,
                              const float* __restrict__ xl,
                              float* __restrict__ den, float* __restrict__ agg,
                              long long E_) {
    long long e = (long long)blockIdx.x * (blockDim.x >> 5) + (threadIdx.x >> 5);
    if (e >= E_) return;
    int lane = threadIdx.x & 31;
    int sN = src[e], dN = dst[e];
    float p[4];
#pragma unroll
    for (int h = 0; h < 4; ++h)
        p[h] = __expf(s_in[e * 4 + h] - fdecode(mx[(long long)dN * 4 + h]));
    if (lane < 4) atomicAdd(&den[(long long)dN * 4 + lane], p[lane]);
#pragma unroll
    for (int h = 0; h < 4; ++h) {
        int c = h * 32 + lane;
        atomicAdd(&agg[(long long)dN * 128 + c], p[h] * xl[(long long)sN * 128 + c]);
    }
}

// ---------------------------------------------------------------------------
// Node finish: normalize, ELU, residual, LayerNorm; emit fp32 (+ optional bf16)
// ---------------------------------------------------------------------------
__global__ void node_finish(const float* __restrict__ agg, const float* __restrict__ den,
                            const float* __restrict__ cb, const float* __restrict__ res,
                            const float* __restrict__ g, const float* __restrict__ be,
                            float* __restrict__ hout, __bf16* __restrict__ hbout, int N_) {
    int n = blockIdx.x * (blockDim.x >> 5) + (threadIdx.x >> 5);
    if (n >= N_) return;
    int lane = threadIdx.x & 31;
    float z[4], s1 = 0.0f, s2 = 0.0f;
#pragma unroll
    for (int h = 0; h < 4; ++h) {
        int c = h * 32 + lane;
        float v = agg[(long long)n * 128 + c] / (den[(long long)n * 4 + h] + 1e-16f);
        v += cb[c];
        v = (v > 0.0f) ? v : (__expf(v) - 1.0f);   // ELU
        v += res[(long long)n * 128 + c];
        z[h] = v; s1 += v; s2 += v * v;
    }
#pragma unroll
    for (int off = 16; off > 0; off >>= 1) {
        s1 += __shfl_xor(s1, off, 32);
        s2 += __shfl_xor(s2, off, 32);
    }
    float mu = s1 * (1.0f / 128.0f);
    float var = s2 * (1.0f / 128.0f) - mu * mu;
    float rs = rsqrtf(var + 1e-5f);
#pragma unroll
    for (int h = 0; h < 4; ++h) {
        int c = h * 32 + lane;
        float o = (z[h] - mu) * rs * g[c] + be[c];
        hout[(long long)n * 128 + c] = o;
        if (hbout) hbout[(long long)n * 128 + c] = (__bf16)o;
    }
}

// ---------------------------------------------------------------------------
// Layer-1 node GEMM: Out[16x16 tile] = hb0(bf16) @ Wt(bf16, pre-transposed)
// One wave per tile, K=128 -> 4 WMMA steps. Uniform full-tile fast path.
// ---------------------------------------------------------------------------
__global__ __launch_bounds__(32)
void gemm_nodes(const __bf16* __restrict__ A, const __bf16* __restrict__ Wt,
                const float* __restrict__ bias, float* __restrict__ Out, int N_) {
    int tile = blockIdx.x;
    int ct = blockIdx.y;          // column tile 0..7
    int lane = threadIdx.x & 31;
    int m = lane & 15, kh = lane >> 4;
    bool full = (tile * 16 + 16 <= N_);
    int mrow = tile * 16 + m;
    if (mrow >= N_) mrow = N_ - 1;
    int col = ct * 16 + m;

    v8f acc = {};
#pragma unroll
    for (int t = 0; t < 4; ++t) {
        const __bf16* ap = A + (long long)mrow * 128 + t * 32 + kh * 8;
        v16bf a = ldfrag2(ap, ap + 16);
        const __bf16* bp = Wt + (long long)col * 128 + t * 32 + kh * 8;
        v16bf b = ldfrag2(bp, bp + 16);
        acc = wmma_bf16(a, b, acc);
    }
    float bv = bias[col];
    if (full) {
#pragma unroll
        for (int r = 0; r < 8; ++r) {
            int row = tile * 16 + r + kh * 8;
            Out[(long long)row * 128 + col] = acc[r] + bv;
        }
    } else {
#pragma unroll
        for (int r = 0; r < 8; ++r) {
            int row = tile * 16 + r + kh * 8;
            if (row < N_) Out[(long long)row * 128 + col] = acc[r] + bv;
        }
    }
}

// ---------------------------------------------------------------------------
// Edge MLP: feat(16 edges x 544) @ Dw1t -> relu -> @ Dw2t -> relu -> @ Dw3
// Wave per 16-edge tile; 2 waves per block. 35 WMMAs per wave.
// ---------------------------------------------------------------------------
__global__ __launch_bounds__(64)
void mlp_edges(const int* __restrict__ src, const int* __restrict__ dst,
               const float* __restrict__ ea, const float* __restrict__ h1,
               const __bf16* __restrict__ W1t, const float* __restrict__ Db1,
               const __bf16* __restrict__ W2t, const float* __restrict__ Db2,
               const float* __restrict__ Dw3, const float* __restrict__ Db3,
               float* __restrict__ out, long long E_, long long Etiles) {
    __shared__ __align__(16) __bf16 feat[2][16][FK];
    __shared__ __align__(16) __bf16 z1s[2][16][32];

    int w = threadIdx.x >> 5;
    int lane = threadIdx.x & 31;
    long long tileIdx = (long long)blockIdx.x * 2 + w;
    bool active = tileIdx < Etiles;
    long long e0 = tileIdx * 16;

    if (active) {
        // Build the 16 x 544 bf16 feature tile in LDS.
        // Each lane: one float4 of hs + one of hd covers all 128 channels/row.
        for (int r = 0; r < 16; ++r) {
            long long e = e0 + r; if (e >= E_) e = E_ - 1;
            long long sN = src[e], dN = dst[e];
            v4f a = *(const v4f*)(h1 + sN * 128 + lane * 4);
            v4f b = *(const v4f*)(h1 + dN * 128 + lane * 4);
            v4bf fa, fb, fd, fp;
#pragma unroll
            for (int q = 0; q < 4; ++q) {
                fa[q] = (__bf16)a[q];
                fb[q] = (__bf16)b[q];
                fd[q] = (__bf16)fabsf(a[q] - b[q]);
                fp[q] = (__bf16)(a[q] * b[q]);
            }
            *(v4bf*)&feat[w][r][lane * 4]       = fa;   // hs
            *(v4bf*)&feat[w][r][128 + lane * 4] = fb;   // hd
            *(v4bf*)&feat[w][r][256 + lane * 4] = fd;   // |hs-hd|
            *(v4bf*)&feat[w][r][384 + lane * 4] = fp;   // hs*hd
            // tail: 6 edge attrs + zero padding up to 544 (lanes 0..31 -> 512..543)
            feat[w][r][512 + lane] = (lane < 6) ? (__bf16)ea[e * 6 + lane] : (__bf16)0.0f;
        }
    }
    __syncthreads();

    int m = lane & 15, kh = lane >> 4;
    if (active) {
        // GEMM1: 16x544 @ 544x32 via 17 K-steps, two 16-wide column tiles.
        v8f acc0 = {}, acc1 = {};
#pragma unroll 4
        for (int t = 0; t < 17; ++t) {
            const __bf16* ap = &feat[w][m][t * 32 + kh * 8];
            v16bf a = ldfrag2(ap, ap + 16);
            const __bf16* b0 = W1t + (long long)m * FK + t * 32 + kh * 8;
            const __bf16* b1 = W1t + (long long)(m + 16) * FK + t * 32 + kh * 8;
            v16bf bf0 = ldfrag2(b0, b0 + 16);
            v16bf bf1 = ldfrag2(b1, b1 + 16);
            acc0 = wmma_bf16(a, bf0, acc0);
            acc1 = wmma_bf16(a, bf1, acc1);
        }
        float bb0 = Db1[m], bb1 = Db1[m + 16];
#pragma unroll
        for (int r = 0; r < 8; ++r) {
            int row = r + kh * 8;
            float v0 = acc0[r] + bb0; v0 = v0 > 0.0f ? v0 : 0.0f;
            float v1 = acc1[r] + bb1; v1 = v1 > 0.0f ? v1 : 0.0f;
            z1s[w][row][m]      = (__bf16)v0;
            z1s[w][row][m + 16] = (__bf16)v1;
        }
    }
    __syncthreads();

    if (active) {
        // GEMM2: 16x32 @ 32x16 (one WMMA), relu, then 16->1 scalar dot.
        const __bf16* ap = &z1s[w][m][kh * 8];
        v16bf a = ldfrag2(ap, ap + 16);
        const __bf16* bp = W2t + m * 32 + kh * 8;
        v16bf b = ldfrag2(bp, bp + 16);
        v8f acc = {};
        acc = wmma_bf16(a, b, acc);

        float b2 = Db2[m];
        float w3 = Dw3[m];
        float b3 = Db3[0];
#pragma unroll
        for (int r = 0; r < 8; ++r) {
            float v = acc[r] + b2; v = v > 0.0f ? v : 0.0f;
            float part = v * w3;
#pragma unroll
            for (int off = 1; off < 16; off <<= 1) part += __shfl_xor(part, off, 32);
            if (m == 0) {
                long long eo = e0 + r + kh * 8;
                if (eo < E_) out[eo] = part + b3;
            }
        }
    }
}

// ---------------------------------------------------------------------------
// Host launch
// ---------------------------------------------------------------------------
extern "C" void kernel_launch(void* const* d_in, const int* in_sizes, int n_in,
                              void* d_out, int out_size, void* d_ws, size_t ws_size,
                              hipStream_t stream) {
    const float* x    = (const float*)d_in[0];
    const int*   ei   = (const int*)d_in[1];
    const float* ea   = (const float*)d_in[2];
    const float* Wl0  = (const float*)d_in[3];
    const float* bl0  = (const float*)d_in[4];
    const float* Wr0  = (const float*)d_in[5];
    const float* br0  = (const float*)d_in[6];
    const float* We0  = (const float*)d_in[7];
    const float* att0 = (const float*)d_in[8];
    const float* cb0  = (const float*)d_in[9];
    const float* g0   = (const float*)d_in[10];
    const float* be0  = (const float*)d_in[11];
    const float* Rw0  = (const float*)d_in[12];
    const float* Rb0  = (const float*)d_in[13];
    const float* Wl1  = (const float*)d_in[14];
    const float* bl1  = (const float*)d_in[15];
    const float* Wr1  = (const float*)d_in[16];
    const float* br1  = (const float*)d_in[17];
    const float* We1  = (const float*)d_in[18];
    const float* att1 = (const float*)d_in[19];
    const float* cb1  = (const float*)d_in[20];
    const float* g1   = (const float*)d_in[21];
    const float* be1  = (const float*)d_in[22];
    const float* Dw1  = (const float*)d_in[23];
    const float* Db1  = (const float*)d_in[24];
    const float* Dw2  = (const float*)d_in[25];
    const float* Db2  = (const float*)d_in[26];
    const float* Dw3  = (const float*)d_in[27];
    const float* Db3  = (const float*)d_in[28];

    const int N = in_sizes[0] / 3;
    const long long E = (long long)in_sizes[2] / 6;
    const int* srcI = ei;
    const int* dstI = ei + E;
    float* out = (float*)d_out;

    // Workspace layout (all offsets 16B-aligned)
    const size_t NB = (size_t)N * 128;
    char* ws = (char*)d_ws;
    float* xl   = (float*)ws;                       // N*128
    float* xr   = xl + NB;                          // N*128
    float* res  = xr + NB;                          // N*128 (later reused as h1)
    float* agg  = res + NB;                         // N*128
    float* h0   = agg + NB;                         // N*128
    float* sbuf = h0 + NB;                          // E*4
    float* den  = sbuf + (size_t)E * 4;             // N*4
    unsigned* mx = (unsigned*)(den + (size_t)N * 4);// N*4
    __bf16* hb0  = (__bf16*)(mx + (size_t)N * 4);   // N*128 bf16
    __bf16* W1t  = hb0 + NB;                        // 32 x 544
    __bf16* W2t  = W1t + (size_t)32 * FK;           // 16 x 32
    __bf16* Wl1t = W2t + (size_t)16 * 32;           // 128 x 128
    __bf16* Wr1t = Wl1t + (size_t)128 * 128;        // 128 x 128
    float* h1 = res;                                // reuse res buffer for h1

    const int B = 256;
    const int nGridNC = (int)((NB + B - 1) / B);
    const int nGridE  = (int)((E + 7) / 8);         // 8 edges per 256-thread block
    const int nGridN  = (N + 7) / 8;                // 8 nodes per 256-thread block

    // --- weight prep (bf16 transposed) ---
    tr_bf16<<<(32 * FK + B - 1) / B, B, 0, stream>>>(Dw1, 518, 32, W1t, FK);
    tr_bf16<<<(16 * 32 + B - 1) / B, B, 0, stream>>>(Dw2, 32, 16, W2t, 32);
    tr_bf16<<<(128 * 128 + B - 1) / B, B, 0, stream>>>(Wl1, 128, 128, Wl1t, 128);
    tr_bf16<<<(128 * 128 + B - 1) / B, B, 0, stream>>>(Wr1, 128, 128, Wr1t, 128);

    // --- GAT block 0 ---
    lin0<<<nGridNC, B, 0, stream>>>(x, Wl0, bl0, Wr0, br0, Rw0, Rb0, xl, xr, res, N);
    init_seg<<<nGridNC, B, 0, stream>>>(den, mx, agg, N);
    edge_score<<<nGridE, B, 0, stream>>>(srcI, dstI, ea, xl, xr, We0, att0, sbuf, mx, E);
    edge_soft_agg<<<nGridE, B, 0, stream>>>(srcI, dstI, sbuf, mx, xl, den, agg, E);
    node_finish<<<nGridN, B, 0, stream>>>(agg, den, cb0, res, g0, be0, h0, hb0, N);

    // --- GAT block 1 (WMMA node transforms, identity residual) ---
    dim3 gGemm((N + 15) / 16, 8);
    gemm_nodes<<<gGemm, 32, 0, stream>>>(hb0, Wl1t, bl1, xl, N);
    gemm_nodes<<<gGemm, 32, 0, stream>>>(hb0, Wr1t, br1, xr, N);
    init_seg<<<nGridNC, B, 0, stream>>>(den, mx, agg, N);
    edge_score<<<nGridE, B, 0, stream>>>(srcI, dstI, ea, xl, xr, We1, att1, sbuf, mx, E);
    edge_soft_agg<<<nGridE, B, 0, stream>>>(srcI, dstI, sbuf, mx, xl, den, agg, E);
    node_finish<<<nGridN, B, 0, stream>>>(agg, den, cb1, h0, g1, be1, h1, (__bf16*)nullptr, N);

    // --- Edge MLP (WMMA) ---
    long long Etiles = (E + 15) / 16;
    int mlpBlocks = (int)((Etiles + 1) / 2);
    mlp_edges<<<mlpBlocks, 64, 0, stream>>>(srcI, dstI, ea, h1,
                                            W1t, Db1, W2t, Db2, Dw3, Db3,
                                            out, E, Etiles);
}
